// APIMixer_26474178412981
// MI455X (gfx1250) — matrix operand order repaired and checked by reading, not compiled, not run on hardware
//
#include <hip/hip_runtime.h>
#include <hip/hip_bf16.h>

typedef __attribute__((ext_vector_type(16))) _Float16 v16h;
typedef __attribute__((ext_vector_type(8)))  _Float16 v8h;
typedef __attribute__((ext_vector_type(2)))  _Float16 v2h;
typedef __attribute__((ext_vector_type(8)))  float    v8f;

#define BT 16384           // 64 * 256
#define STATE_DIM 224

__device__ __forceinline__ v16h join16(v8h lo, v8h hi) {
    return __builtin_shufflevector(lo, hi, 0,1,2,3,4,5,6,7,8,9,10,11,12,13,14,15);
}

// ---------------------------------------------------------------------------
// Hypernet: out_mean[s,o] = mean_a( x @ W(x) ), recast as GEMM
//   G[r, k*DIN+i] = h[r,k]*x[r,i],  W2r[k*DIN+i, o] = h2w[k, i*64+o]
// One wave = 16-row M-tile = 2 samples. K-chunk = 16 hyper-hidden k's.
// Fragments are fed by ds_load_b128 from pre-swizzled LDS.
// ---------------------------------------------------------------------------
template<int DIN>
__global__ __launch_bounds__(128)
void hyper_kernel(const float* __restrict__ states,
                  const float* __restrict__ h1w,   // DIN x 64
                  const float* __restrict__ h1b,   // 64
                  const float* __restrict__ h2w,   // 64 x (DIN*64)
                  const float* __restrict__ h2b,   // DIN*64
                  float* __restrict__ out_mean,    // BT x 64
                  int state_off)
{
    constexpr int K   = DIN * 64;      // 512 / 384
    constexpr int KC  = DIN * 16;      // 128 / 96  (16 k's per chunk)
    constexpr int KCP = KC + 8;        // padded row stride (f16), 16B-aligned
    constexpr int NCH = K / KC;        // 4
    constexpr int KS  = KC / 32;       // k-steps per chunk: 4 / 3

    __shared__ alignas(16) _Float16 BshT[64 * KCP];       // W2r^T chunk
    __shared__ alignas(16) _Float16 Gsh[4][16 * KCP];     // G chunk per wave
    __shared__ float Hsh[4][16 * 68];                     // relu hidden (padded)
    __shared__ float Xsh[4][16 * DIN];                    // inputs

    const int tid  = threadIdx.x;
    const int wave = tid >> 5;
    const int lane = tid & 31;
    const int row0 = (blockIdx.x * 4 + wave) * 16;
    const int mrow = lane & 15;
    const int half = lane >> 4;

    // ---- load x rows (f32) ----
    for (int idx = lane; idx < 16 * DIN; idx += 32) {
        int m = idx / DIN, i = idx % DIN;
        int r = row0 + m;
        Xsh[wave][m * DIN + i] = states[(r >> 3) * STATE_DIM + state_off + (r & 7) * DIN + i];
    }
    __syncthreads();

    // ---- h = relu(x @ h1w + h1b) ----
    for (int idx = lane; idx < 16 * 64; idx += 32) {
        int m = idx >> 6, k = idx & 63;
        float acc = h1b[k];
#pragma unroll
        for (int i = 0; i < DIN; ++i)
            acc += Xsh[wave][m * DIN + i] * h1w[i * 64 + k];
        Hsh[wave][m * 68 + k] = acc > 0.f ? acc : 0.f;
    }
    __syncthreads();

    // x row for this lane's G-build row (m = mrow)
    float xr[DIN];
#pragma unroll
    for (int i = 0; i < DIN; ++i) xr[i] = Xsh[wave][mrow * DIN + i];

    v8f acc[4];
#pragma unroll
    for (int nt = 0; nt < 4; ++nt)
#pragma unroll
        for (int j = 0; j < 8; ++j) acc[nt][j] = 0.f;

    for (int kc = 0; kc < NCH; ++kc) {
        __syncthreads();
        // ---- stage W2r^T chunk (cols x kk), coalesced global reads ----
        {
            const int base = kc * 16 * (DIN * 64);
            for (int idx = tid; idx < KC * 64; idx += 128) {
                int g   = base + idx;
                int k   = g / (DIN * 64);
                int rem = g % (DIN * 64);
                int i   = rem >> 6;
                int col = rem & 63;
                BshT[col * KCP + (k - kc * 16) * DIN + i] = (_Float16)h2w[g];
            }
        }
        // ---- build G chunk: lane owns row mrow, k = kc*16 + half*8 + kb ----
        {
            _Float16* grow = &Gsh[wave][mrow * KCP];
#pragma unroll
            for (int kb = 0; kb < 8; ++kb) {
                int kl = half * 8 + kb;                 // 0..15 within chunk
                float hval = Hsh[wave][mrow * 68 + kc * 16 + kl];
                if constexpr (DIN == 8) {
                    v8h g;
#pragma unroll
                    for (int i = 0; i < 8; ++i) g[i] = (_Float16)(xr[i] * hval);
                    *(v8h*)(grow + kl * 8) = g;
                } else {
                    _Float16* gp = grow + kl * DIN;
#pragma unroll
                    for (int i = 0; i < DIN; i += 2) {
                        v2h g2;
                        g2[0] = (_Float16)(xr[i] * hval);
                        g2[1] = (_Float16)(xr[i + 1] * hval);
                        *(v2h*)(gp + i) = g2;
                    }
                }
            }
        }
        __syncthreads();

        // ---- WMMA: fragments via ds_load_b128 ----
        const _Float16* gbase = &Gsh[wave][mrow * KCP];
#pragma unroll
        for (int ks = 0; ks < KS; ++ks) {
            v8h alo = *(const v8h*)(gbase + ks * 32 + half * 8);
            v8h ahi = *(const v8h*)(gbase + ks * 32 + 16 + half * 8);
            v16h afrag = join16(alo, ahi);
#pragma unroll
            for (int nt = 0; nt < 4; ++nt) {
                const _Float16* bbase = &BshT[(nt * 16 + mrow) * KCP + ks * 32 + half * 8];
                v16h bfrag = join16(*(const v8h*)(bbase), *(const v8h*)(bbase + 16));
                acc[nt] = __builtin_amdgcn_wmma_f32_16x16x32_f16(
                    false, afrag, false, bfrag, (short)0, acc[nt], false, false);
            }
        }
    }

    // ---- epilogue: mean over 8 agents (sum of the 8 C-VGPRs) + h2b term ----
    int s = (row0 >> 3) + half;
    float xsum[DIN];
#pragma unroll
    for (int i = 0; i < DIN; ++i) {
        float t = 0.f;
#pragma unroll
        for (int a = 0; a < 8; ++a) t += Xsh[wave][(half * 8 + a) * DIN + i];
        xsum[i] = t;
    }
#pragma unroll
    for (int nt = 0; nt < 4; ++nt) {
        int col = nt * 16 + mrow;
        float ssum = 0.f;
#pragma unroll
        for (int j = 0; j < 8; ++j) ssum += acc[nt][j];
        float hb = 0.f;
#pragma unroll
        for (int i = 0; i < DIN; ++i) hb += xsum[i] * h2b[i * 64 + col];
        out_mean[s * 64 + col] = (ssum + hb) * 0.125f;
    }
}

// ---------------------------------------------------------------------------
// se = relu(ally_mean + enemy_mean + mean_a(actions) @ act_w + biases)
// ---------------------------------------------------------------------------
__global__ __launch_bounds__(256)
void se_kernel(const float* __restrict__ states,
               const float* __restrict__ ally_m, const float* __restrict__ enemy_m,
               const float* __restrict__ act_w, const float* __restrict__ act_b,
               const float* __restrict__ al_bias, const float* __restrict__ en_bias,
               float* __restrict__ se)
{
    __shared__ float abar[4][14];
    int tid = threadIdx.x;
    int sl  = tid >> 6;
    int col = tid & 63;
    int s   = blockIdx.x * 4 + sl;
    if (col < 14) {
        float t = 0.f;
#pragma unroll
        for (int a = 0; a < 8; ++a) t += states[s * STATE_DIM + 112 + a * 14 + col];
        abar[sl][col] = t * 0.125f;
    }
    __syncthreads();
    float e = 0.f;
#pragma unroll
    for (int j = 0; j < 14; ++j) e += abar[sl][j] * act_w[j * 64 + col];
    float v = ally_m[s * 64 + col] + enemy_m[s * 64 + col] + e
            + act_b[col] + al_bias[col] + en_bias[col];
    se[s * 64 + col] = v > 0.f ? v : 0.f;
}

// ---------------------------------------------------------------------------
// Final mix: w1 = softmax_a(hidden_states @ hw1_w) via WMMA (b128-fed frags),
// then y = elu(q@w1 + b1) @ |se@hw2_w| + se@hb2_w. One wave = 2 samples.
// ---------------------------------------------------------------------------
__global__ __launch_bounds__(128)
void mix_kernel(const float* __restrict__ qvals, const float* __restrict__ hstates,
                const float* __restrict__ se,
                const float* __restrict__ hw1_w, const float* __restrict__ hw1_b,
                const float* __restrict__ hb1_w, const float* __restrict__ hb1_b,
                const float* __restrict__ hw2_w, const float* __restrict__ hw2_b,
                const float* __restrict__ hb2_w, const float* __restrict__ hb2_b,
                float* __restrict__ out)
{
    __shared__ alignas(16) _Float16 BshT[32 * 72];     // hw1_w^T, padded
    __shared__ alignas(16) _Float16 Hs[4][16 * 72];    // hidden rows f16, padded
    __shared__ float Ssh[4][2 * 64];                   // se rows per wave

    int tid = threadIdx.x, wave = tid >> 5, lane = tid & 31;
    int row0 = (blockIdx.x * 4 + wave) * 16;
    int mrow = lane & 15, half = lane >> 4;
    int s = (row0 >> 3) + half;

    for (int idx = tid; idx < 64 * 32; idx += 128) {
        int k = idx >> 5, e = idx & 31;
        BshT[e * 72 + k] = (_Float16)hw1_w[idx];
    }
    for (int idx = lane; idx < 16 * 64; idx += 32) {
        Hs[wave][(idx >> 6) * 72 + (idx & 63)] = (_Float16)hstates[row0 * 64 + idx];
    }
    for (int idx = lane; idx < 128; idx += 32) {
        Ssh[wave][idx] = se[(row0 >> 3) * 64 + idx];
    }
    __syncthreads();

    v8f acc[2];
#pragma unroll
    for (int nt = 0; nt < 2; ++nt)
#pragma unroll
        for (int j = 0; j < 8; ++j) acc[nt][j] = 0.f;

    const _Float16* gbase = &Hs[wave][mrow * 72];
#pragma unroll
    for (int ks = 0; ks < 2; ++ks) {       // K = 64
        v16h afrag = join16(*(const v8h*)(gbase + ks * 32 + half * 8),
                            *(const v8h*)(gbase + ks * 32 + 16 + half * 8));
#pragma unroll
        for (int nt = 0; nt < 2; ++nt) {   // N = 32
            const _Float16* bbase = &BshT[(nt * 16 + mrow) * 72 + ks * 32 + half * 8];
            v16h bfrag = join16(*(const v8h*)(bbase), *(const v8h*)(bbase + 16));
            acc[nt] = __builtin_amdgcn_wmma_f32_16x16x32_f16(
                false, afrag, false, bfrag, (short)0, acc[nt], false, false);
        }
    }

    // softmax over agents (the 8 C-VGPRs) per (lane, n-tile)
    float w1v[2][8];
#pragma unroll
    for (int nt = 0; nt < 2; ++nt) {
        int e = nt * 16 + mrow;
        float bias = hw1_b[e];
        float mx = -1e30f;
#pragma unroll
        for (int j = 0; j < 8; ++j) {
            float l = acc[nt][j] + bias;
            w1v[nt][j] = l;
            mx = l > mx ? l : mx;
        }
        float ssum = 0.f;
#pragma unroll
        for (int j = 0; j < 8; ++j) {
            float ex = __expf(w1v[nt][j] - mx);
            w1v[nt][j] = ex;
            ssum += ex;
        }
        float inv = 1.f / ssum;
#pragma unroll
        for (int j = 0; j < 8; ++j) w1v[nt][j] *= inv;
    }

    float qv[8];
#pragma unroll
    for (int a = 0; a < 8; ++a) qv[a] = qvals[s * 8 + a];

    const float* srow = &Ssh[wave][half * 64];
    float b2 = hb2_b[0];
    for (int k = 0; k < 64; ++k) b2 += srow[k] * hb2_w[k];

    float ypart = 0.f;
#pragma unroll
    for (int nt = 0; nt < 2; ++nt) {
        int e = nt * 16 + mrow;
        float b1 = hb1_b[e], w2 = hw2_b[e];
        for (int k = 0; k < 64; ++k) {
            float sk = srow[k];
            b1 += sk * hb1_w[k * 32 + e];
            w2 += sk * hw2_w[k * 32 + e];
        }
        w2 = fabsf(w2);
        float hpre = b1;
#pragma unroll
        for (int j = 0; j < 8; ++j) hpre += qv[j] * w1v[nt][j];
        float h = hpre > 0.f ? hpre : (__expf(hpre) - 1.f);   // elu, alpha=1
        ypart += h * w2;
    }

    ypart += __shfl_xor(ypart, 1);
    ypart += __shfl_xor(ypart, 2);
    ypart += __shfl_xor(ypart, 4);
    ypart += __shfl_xor(ypart, 8);
    if (mrow == 0) out[s] = ypart + b2;
}

extern "C" void kernel_launch(void* const* d_in, const int* in_sizes, int n_in,
                              void* d_out, int out_size, void* d_ws, size_t ws_size,
                              hipStream_t stream) {
    const float* qvals   = (const float*)d_in[0];
    const float* states  = (const float*)d_in[1];
    const float* hstates = (const float*)d_in[2];
    const float* hw1_w   = (const float*)d_in[3];
    const float* hw1_b   = (const float*)d_in[4];
    const float* en_h1w  = (const float*)d_in[5];
    const float* en_h1b  = (const float*)d_in[6];
    const float* en_h2w  = (const float*)d_in[7];
    const float* en_h2b  = (const float*)d_in[8];
    const float* en_bias = (const float*)d_in[9];
    const float* al_h1w  = (const float*)d_in[10];
    const float* al_h1b  = (const float*)d_in[11];
    const float* al_h2w  = (const float*)d_in[12];
    const float* al_h2b  = (const float*)d_in[13];
    const float* al_bias = (const float*)d_in[14];
    const float* act_w   = (const float*)d_in[15];
    const float* act_b   = (const float*)d_in[16];
    const float* hb1_w   = (const float*)d_in[17];
    const float* hb1_b   = (const float*)d_in[18];
    const float* hw2_w   = (const float*)d_in[19];
    const float* hw2_b   = (const float*)d_in[20];
    const float* hb2_w   = (const float*)d_in[21];
    const float* hb2_b   = (const float*)d_in[22];

    float* ally_m  = (float*)d_ws;                 // BT*64 f32
    float* enemy_m = ally_m  + (size_t)BT * 64;    // BT*64 f32
    float* se_buf  = enemy_m + (size_t)BT * 64;    // BT*64 f32

    hyper_kernel<8><<<2048, 128, 0, stream>>>(states, al_h1w, al_h1b, al_h2w, al_h2b,
                                              ally_m, 0);
    hyper_kernel<6><<<2048, 128, 0, stream>>>(states, en_h1w, en_h1b, en_h2w, en_h2b,
                                              enemy_m, 64);
    se_kernel<<<BT / 4, 256, 0, stream>>>(states, ally_m, enemy_m, act_w, act_b,
                                          al_bias, en_bias, se_buf);
    mix_kernel<<<2048, 128, 0, stream>>>(qvals, hstates, se_buf,
                                         hw1_w, hw1_b, hb1_w, hb1_b,
                                         hw2_w, hw2_b, hb2_w, hb2_b,
                                         (float*)d_out);
}
